// NemotronHMOE_12481174962825
// MI455X (gfx1250) — matrix operand order repaired and checked by reading, not compiled, not run on hardware
//
#include <hip/hip_runtime.h>

// Problem constants
#define T_TOK 2048
#define H_DIM 2048
#define LATD  1024
#define INTERD 1024
#define NEXP  32
#define SH_I  4096

typedef __attribute__((ext_vector_type(16))) __bf16 v16bf;
typedef __attribute__((ext_vector_type(8)))  __bf16 bf16x8;
typedef __attribute__((ext_vector_type(8)))  float  v8f;
typedef __attribute__((ext_vector_type(4)))  float  f32x4;

// Native f32 -> bf16 convert (RNE; lowers to v_cvt_pk_bf16_f32 when paired).
__device__ __forceinline__ __bf16 f2bf(float f) { return (__bf16)f; }

__device__ __forceinline__ v8f wmma_bf16(v16bf a, v16bf b, v8f c) {
  // D = A(16x32) x B(32x16) + C, f32 accumulate
  return __builtin_amdgcn_wmma_f32_16x16x32_bf16(false, a, false, b, (short)0, c,
                                                 false, false);
}

// Async DMA: per-lane 16B global -> LDS copy, tracked by ASYNCcnt.
__device__ __forceinline__ void async_g2l_b128(void* lds_dst, const void* gsrc) {
  unsigned lds_addr = (unsigned)(unsigned long long)(uintptr_t)lds_dst;
  asm volatile("global_load_async_to_lds_b128 %0, %1, off"
               :: "v"(lds_addr), "v"(gsrc) : "memory");
}
__device__ __forceinline__ void wait_asynccnt0() {
  asm volatile("s_wait_asynccnt 0x0" ::: "memory");
}

// -------- shared-memory block for the tiled GEMM core --------
struct __align__(16) SMem {
  __bf16 a[128 * 40];        // A tile 128 x 32, stride 40 (80B rows, 16B aligned)
  __bf16 b[128 * 40];        // B tile 128(N) x 32(K)
  long   aoff[128];          // per-row A element offsets (gather support)
  int    ridx[128];          // token or h-slot per row
  float  rw[128];            // per-row combine weight
  int    rvalid[128];        // row validity
};

__device__ __forceinline__ void cvt_store8(__bf16* dst, f32x4 q0, f32x4 q1) {
  bf16x8 v;
  v[0] = f2bf(q0[0]); v[1] = f2bf(q0[1]); v[2] = f2bf(q0[2]); v[3] = f2bf(q0[3]);
  v[4] = f2bf(q1[0]); v[5] = f2bf(q1[1]); v[6] = f2bf(q1[2]); v[7] = f2bf(q1[3]);
  *(bf16x8*)dst = v;
}

// 128x128 block tile, 8 waves (2x4), each wave 64x32 = 8 wmma accumulators.
// A rows are addressed as Abase + sm.aoff[row] (element offsets), keeping
// global address-space provenance so loads lower to global_load_b128; when A
// is already bf16 the tile is staged with async global->LDS DMA (ASYNCcnt).
template<bool ABF16>
__device__ __forceinline__ void gemm_core(SMem& sm, const void* __restrict__ Abase,
                                          const float* __restrict__ B,
                                          long ldb, long n0, int K, v8f (&acc)[4][2]) {
  const int t    = threadIdx.x;
  const int lane = t & 31;
  const int wid  = t >> 5;
  const int wm   = (wid & 1) << 6;   // wave M offset: 0 / 64
  const int wn   = (wid >> 1) << 5;  // wave N offset: 0..96
  const int lrow = t >> 1;           // staging: row 0..127
  const int lseg = (t & 1) << 4;     // staging: k sub-offset 0 / 16
  const int am  = lane & 15;         // A frag row within subtile
  const int ah  = (lane >> 4) << 3;  // A frag K chunk base: 0 / 8
  const int bn  = lane & 15;         // B frag col
  const int bks = (lane >> 4) << 3;  // B frag K chunk base: 0 / 8

  // hoisted, loop-invariant source pointers (single LDS read for the offset)
  const long aoff = sm.aoff[lrow];
  const __bf16* asrcb = (const __bf16*)Abase + aoff + lseg;
  const float*  asrcf = (const float*)Abase + aoff + lseg;
  const float*  bsrc  = B + (n0 + lrow) * ldb + lseg;
  __bf16* adst = sm.a + lrow * 40 + lseg;
  __bf16* bdst = sm.b + lrow * 40 + lseg;

#pragma unroll
  for (int mi = 0; mi < 4; ++mi)
#pragma unroll
    for (int ni = 0; ni < 2; ++ni)
#pragma unroll
      for (int r = 0; r < 8; ++r) acc[mi][ni][r] = 0.f;

  for (int k0 = 0; k0 < K; k0 += 32) {
    // ---- stage A tile ----
    if constexpr (ABF16) {
      // raw bf16 copy: async DMA straight into LDS (no VGPR round-trip)
      async_g2l_b128(adst,     asrcb + k0);
      async_g2l_b128(adst + 8, asrcb + k0 + 8);
      if (k0 + 32 < K) __builtin_prefetch(asrcb + k0 + 32, 0, 1);
    } else {
      const float* src = asrcf + k0;
      f32x4 q0 = ((const f32x4*)src)[0];
      f32x4 q1 = ((const f32x4*)src)[1];
      f32x4 q2 = ((const f32x4*)src)[2];
      f32x4 q3 = ((const f32x4*)src)[3];
      cvt_store8(adst,     q0, q1);
      cvt_store8(adst + 8, q2, q3);
      if (k0 + 32 < K) __builtin_prefetch(src + 32, 0, 1);
    }
    // ---- stage B tile (weights, always f32 rows [N][K]: load+cvt+ds_store) ----
    {
      const float* src = bsrc + k0;
      f32x4 q0 = ((const f32x4*)src)[0];
      f32x4 q1 = ((const f32x4*)src)[1];
      f32x4 q2 = ((const f32x4*)src)[2];
      f32x4 q3 = ((const f32x4*)src)[3];
      cvt_store8(bdst,     q0, q1);
      cvt_store8(bdst + 8, q2, q3);
      if (k0 + 32 < K) __builtin_prefetch(src + 32, 0, 1);
    }
    if constexpr (ABF16) wait_asynccnt0();   // A tile DMA complete in LDS
    __syncthreads();

    // ---- B fragments (lane: col = bn, K = bks..bks+7 and 16+bks..16+bks+7) ----
    v16bf bf[2];
#pragma unroll
    for (int ni = 0; ni < 2; ++ni) {
      const __bf16* bp = sm.b + (wn + ni * 16 + bn) * 40 + bks;
      bf16x8 lo = *(const bf16x8*)bp;
      bf16x8 hi = *(const bf16x8*)(bp + 16);
      bf[ni] = __builtin_shufflevector(lo, hi, 0, 1, 2, 3, 4, 5, 6, 7,
                                       8, 9, 10, 11, 12, 13, 14, 15);
    }
    // ---- A fragments + WMMA ----
#pragma unroll
    for (int mi = 0; mi < 4; ++mi) {
      const __bf16* ap = sm.a + (wm + mi * 16 + am) * 40;
      bf16x8 lo = *(const bf16x8*)(ap + ah);          // K = 8h .. 8h+7
      bf16x8 hi = *(const bf16x8*)(ap + 16 + ah);     // K = 16+8h .. 16+8h+7
      v16bf af = __builtin_shufflevector(lo, hi, 0, 1, 2, 3, 4, 5, 6, 7,
                                         8, 9, 10, 11, 12, 13, 14, 15);
      acc[mi][0] = wmma_bf16(af, bf[0], acc[mi][0]);
      acc[mi][1] = wmma_bf16(af, bf[1], acc[mi][1]);
    }
    __syncthreads();
  }
}

// Visit every accumulator element as (row, col, value) in block-tile coords.
template<class F>
__device__ __forceinline__ void for_each_acc(v8f (&acc)[4][2], F&& f) {
  const int lane = threadIdx.x & 31;
  const int wid  = threadIdx.x >> 5;
  const int wm = (wid & 1) << 6, wn = (wid >> 1) << 5;
  const int rr = (lane >> 4) << 3, cc = lane & 15;
#pragma unroll
  for (int mi = 0; mi < 4; ++mi)
#pragma unroll
    for (int ni = 0; ni < 2; ++ni)
#pragma unroll
      for (int r = 0; r < 8; ++r)
        f(wm + mi * 16 + rr + r, wn + ni * 16 + cc, acc[mi][ni][r]);
}

enum { EP_BF16 = 0, EP_RELU2_BF16 = 1, EP_F32 = 2, EP_ADDF32 = 3 };

template<bool ABF16, int EP>
__global__ __launch_bounds__(256) void k_gemm(const void* __restrict__ A, long lda,
                                              const float* __restrict__ B, long ldb,
                                              int K, void* __restrict__ C, long ldc) {
  __shared__ SMem sm;
  const long m0 = (long)blockIdx.x * 128;
  const long n0 = (long)blockIdx.y * 128;
  if (threadIdx.x < 128) sm.aoff[threadIdx.x] = (m0 + threadIdx.x) * lda;
  __syncthreads();
  v8f acc[4][2];
  gemm_core<ABF16>(sm, A, B, ldb, n0, K, acc);
  for_each_acc(acc, [&](int row, int col, float v) {
    long idx = (m0 + row) * ldc + (n0 + col);
    if constexpr (EP == EP_BF16) {
      ((__bf16*)C)[idx] = f2bf(v);
    } else if constexpr (EP == EP_RELU2_BF16) {
      float r = fmaxf(v, 0.f);
      ((__bf16*)C)[idx] = f2bf(r * r);
    } else if constexpr (EP == EP_F32) {
      ((float*)C)[idx] = v;
    } else {
      ((float*)C)[idx] += v;
    }
  });
}

// -------- router: gate logits + noaux_tc group routing + bucket build --------
__global__ __launch_bounds__(256) void k_route(const float* __restrict__ x,
                                               const float* __restrict__ gw,
                                               const float* __restrict__ gb,
                                               int* __restrict__ counts,
                                               int* __restrict__ btok,
                                               int* __restrict__ bslot,
                                               float* __restrict__ bw) {
  __shared__ float sc[8][33];
  const int wid = threadIdx.x >> 5, lane = threadIdx.x & 31;
  const int tkn = blockIdx.x * 8 + wid;
  const float* xr = x + (long)tkn * H_DIM;
  const float* wr = gw + (long)lane * H_DIM;
  float acc = 0.f;
  for (int i = 0; i < H_DIM; i += 4) {
    f32x4 a = *(const f32x4*)(xr + i);
    f32x4 b = *(const f32x4*)(wr + i);
    acc = fmaf(a[0], b[0], fmaf(a[1], b[1], fmaf(a[2], b[2], fmaf(a[3], b[3], acc))));
  }
  float score = 1.f / (1.f + __expf(-acc));    // sigmoid
  sc[wid][lane] = score;
  __syncthreads();

  // every lane redundantly runs the (tiny, fully unrolled) selection
  float s[32], sb[32];
#pragma unroll
  for (int i = 0; i < 32; ++i) { s[i] = sc[wid][i]; sb[i] = s[i] + gb[i]; }
  float gsc[4];
#pragma unroll
  for (int g = 0; g < 4; ++g) {
    float m1 = -1e30f, m2 = -1e30f;
#pragma unroll
    for (int j = 0; j < 8; ++j) {
      float v = sb[g * 8 + j];
      if (v > m1) { m2 = m1; m1 = v; } else if (v > m2) { m2 = v; }
    }
    gsc[g] = m1 + m2;
  }
  int g0 = 0;
#pragma unroll
  for (int g = 1; g < 4; ++g) if (gsc[g] > gsc[g0]) g0 = g;
  int g1 = -1; float g1v = -1e30f;
#pragma unroll
  for (int g = 0; g < 4; ++g)
    if (g != g0 && gsc[g] > g1v) { g1v = gsc[g]; g1 = g; }

  int sel[8]; float wsum = 0.f;
  bool used[32];
#pragma unroll
  for (int i = 0; i < 32; ++i) used[i] = false;
#pragma unroll
  for (int k = 0; k < 8; ++k) {
    int best = 0; float bv = -1e30f;
#pragma unroll
    for (int i = 0; i < 32; ++i) {
      int g = i >> 3;
      bool ok = (g == g0 || g == g1) && !used[i];
      if (ok && sb[i] > bv) { bv = sb[i]; best = i; }
    }
    used[best] = true; sel[k] = best; wsum += s[best];
  }
  float scale = 2.5f / wsum;

  // lanes 0..7 publish their selected expert (sel[k] with k == lane)
  int myexp = -1;
#pragma unroll
  for (int k = 0; k < 8; ++k) if (lane == k) myexp = sel[k];
  if (lane < 8) {
    int e = myexp;
    int pos = atomicAdd(&counts[e], 1);
    btok [e * T_TOK + pos] = tkn;
    bslot[e * T_TOK + pos] = tkn * 8 + lane;
    bw   [e * T_TOK + pos] = s[e] * scale;
  }
}

// -------- expert up: h = relu2(x_lat @ w1[e]^T) * cw, bf16 to h_ws --------
__global__ __launch_bounds__(256) void k_expert_up(const __bf16* __restrict__ x_lat,
                                                   const float* __restrict__ w1,
                                                   const int* __restrict__ counts,
                                                   const int* __restrict__ btok,
                                                   const int* __restrict__ bslot,
                                                   const float* __restrict__ bw,
                                                   __bf16* __restrict__ h_ws) {
  const int e = blockIdx.z, tile = blockIdx.x;
  const int cnt = counts[e];
  if (tile * 128 >= cnt) return;
  __shared__ SMem sm;
  const int t = threadIdx.x;
  if (t < 128) {
    int g = tile * 128 + t;
    bool valid = g < cnt;
    int tok  = valid ? btok [e * T_TOK + g] : 0;
    int slot = valid ? bslot[e * T_TOK + g] : (T_TOK * 8 + t);  // dummy rows
    float w  = valid ? bw   [e * T_TOK + g] : 0.f;
    sm.aoff[t] = (long)tok * LATD;
    sm.ridx[t] = slot;
    sm.rw[t]   = w;
  }
  __syncthreads();
  v8f acc[4][2];
  const long n0 = (long)blockIdx.y * 128;
  gemm_core<true>(sm, x_lat, w1 + (long)e * INTERD * LATD, LATD, n0, LATD, acc);
  for_each_acc(acc, [&](int row, int col, float v) {
    float w = sm.rw[row];
    long slot = sm.ridx[row];
    float r = fmaxf(v, 0.f);
    h_ws[slot * INTERD + n0 + col] = f2bf(r * r * w);
  });
}

// -------- expert down: y_lat[t] += h @ w2[e]^T (scatter, f32 atomics) --------
__global__ __launch_bounds__(256) void k_expert_down(const __bf16* __restrict__ h_ws,
                                                     const float* __restrict__ w2,
                                                     const int* __restrict__ counts,
                                                     const int* __restrict__ btok,
                                                     const int* __restrict__ bslot,
                                                     float* __restrict__ y_lat) {
  const int e = blockIdx.z, tile = blockIdx.x;
  const int cnt = counts[e];
  if (tile * 128 >= cnt) return;
  __shared__ SMem sm;
  const int t = threadIdx.x;
  if (t < 128) {
    int g = tile * 128 + t;
    bool valid = g < cnt;
    int slot = valid ? bslot[e * T_TOK + g] : (T_TOK * 8 + t);
    int tok  = valid ? btok [e * T_TOK + g] : 0;
    sm.aoff[t]   = (long)slot * INTERD;
    sm.ridx[t]   = tok;
    sm.rvalid[t] = valid ? 1 : 0;
  }
  __syncthreads();
  v8f acc[4][2];
  const long n0 = (long)blockIdx.y * 128;
  gemm_core<true>(sm, h_ws, w2 + (long)e * LATD * INTERD, INTERD, n0, INTERD, acc);
  for_each_acc(acc, [&](int row, int col, float v) {
    if (sm.rvalid[row])
      atomicAdd(&y_lat[(long)sm.ridx[row] * LATD + n0 + col], v);
  });
}

__global__ __launch_bounds__(256) void k_init(float* __restrict__ y_lat,
                                              int* __restrict__ counts) {
  long i = (long)blockIdx.x * 256 + threadIdx.x;
  if (i < (long)T_TOK * LATD) y_lat[i] = 0.f;
  if (i < NEXP) counts[i] = 0;
}

extern "C" void kernel_launch(void* const* d_in, const int* in_sizes, int n_in,
                              void* d_out, int out_size, void* d_ws, size_t ws_size,
                              hipStream_t stream) {
  const float* x   = (const float*)d_in[0];  // [T, H]
  const float* gw  = (const float*)d_in[1];  // [E, H]
  const float* gb  = (const float*)d_in[2];  // [E]
  const float* fc1 = (const float*)d_in[3];  // [LAT, H]
  const float* fc2 = (const float*)d_in[4];  // [H, LAT]
  const float* w1  = (const float*)d_in[5];  // [E, INTER, LAT]
  const float* w2  = (const float*)d_in[6];  // [E, LAT, INTER]
  const float* sup = (const float*)d_in[7];  // [SH_I, H]
  const float* sdn = (const float*)d_in[8];  // [H, SH_I]
  float* out = (float*)d_out;                // [T, H]

  char* ws = (char*)d_ws;
  __bf16* x_lat = (__bf16*)(ws);                 // 2048*1024 bf16
  ws += (size_t)T_TOK * LATD * 2;
  __bf16* h_ws = (__bf16*)(ws);                  // (16384 + 128 dummy) * 1024 bf16
  ws += (size_t)(T_TOK * 8 + 128) * INTERD * 2;
  __bf16* sh_h = (__bf16*)(ws);                  // 2048*4096 bf16
  ws += (size_t)T_TOK * SH_I * 2;
  float* y_lat = (float*)(ws);                   // 2048*1024 f32
  ws += (size_t)T_TOK * LATD * 4;
  int* counts = (int*)(ws); ws += 256;
  int* btok   = (int*)(ws); ws += (size_t)NEXP * T_TOK * 4;
  int* bslot  = (int*)(ws); ws += (size_t)NEXP * T_TOK * 4;
  float* bw   = (float*)(ws);

  // 1) zero y_lat accumulator + bucket counters
  k_init<<<8192, 256, 0, stream>>>(y_lat, counts);
  // 2) gate + routing + per-expert buckets
  k_route<<<T_TOK / 8, 256, 0, stream>>>(x, gw, gb, counts, btok, bslot, bw);
  // 3) latent projection x_lat = x @ fc1^T  (bf16 out)
  k_gemm<false, EP_BF16><<<dim3(16, 8), 256, 0, stream>>>(
      x, H_DIM, fc1, H_DIM, H_DIM, x_lat, LATD);
  // 4) routed experts, sparse
  k_expert_up<<<dim3(16, 8, 32), 256, 0, stream>>>(
      x_lat, w1, counts, btok, bslot, bw, h_ws);
  k_expert_down<<<dim3(16, 8, 32), 256, 0, stream>>>(
      h_ws, w2, counts, btok, bslot, y_lat);
  // 5) shared experts MLP
  k_gemm<false, EP_RELU2_BF16><<<dim3(16, 32), 256, 0, stream>>>(
      x, H_DIM, sup, H_DIM, H_DIM, sh_h, SH_I);
  k_gemm<true, EP_F32><<<dim3(16, 16), 256, 0, stream>>>(
      sh_h, SH_I, sdn, SH_I, SH_I, out, H_DIM);
  // 6) routed output added on top: out += y_lat @ fc2^T
  k_gemm<false, EP_ADDF32><<<dim3(16, 16), 256, 0, stream>>>(
      y_lat, LATD, fc2, LATD, LATD, out, H_DIM);
}
